// GNNAutoencoder_721554505798
// MI455X (gfx1250) — compile-verified, matrix operand order checked by reading, not run on hardware
//
#include <hip/hip_runtime.h>
#include <hip/hip_bf16.h>

// GNN autoencoder (2x GATConv + node/edge decoders) for gfx1250 / MI455X.
// - Dense GEMMs: v_wmma_f32_16x16x32_bf16, bf16 operands resident in global
//   memory (pre-converted once per call), tiles staged to LDS via
//   global_load_async_to_lds_b128 (ASYNCcnt) when available.
// - Attention softmax: atomic segment max/sum/scatter (memory-bound, f32).

typedef __attribute__((ext_vector_type(16))) __bf16 v16bf;
typedef __attribute__((ext_vector_type(8)))  float  v8f;

#if defined(__has_builtin)
#if __has_builtin(__builtin_amdgcn_global_load_async_to_lds_b128)
#define HAVE_ASYNC_LDS 1
#endif
#if __has_builtin(__builtin_amdgcn_s_wait_asynccnt)
#define HAVE_WAIT_ASYNC 1
#endif
#endif

#ifdef HAVE_ASYNC_LDS
// Parameter types per hipcc diagnostic: payload is int __vector_size__(16);
// arg0 = global (AS1 / __device__) pointer, arg1 = LDS (AS3) pointer.
typedef int v4i_b128 __attribute__((__vector_size__(4 * sizeof(int))));
#define ASYNC_G(p) ((__attribute__((address_space(1))) v4i_b128*)(p))
#define ASYNC_L(p) ((__attribute__((address_space(3))) v4i_b128*)(p))
#endif

__device__ __forceinline__ void wait_async_lds() {
#ifdef HAVE_ASYNC_LDS
#ifdef HAVE_WAIT_ASYNC
  __builtin_amdgcn_s_wait_asynccnt(0);
#else
  asm volatile("s_wait_asynccnt 0x0" ::: "memory");
#endif
#endif
}

#define GEMM_STRIDE 40   // bf16 elems per LDS row: 80B, 16B-aligned chunks
#define EMLP_STRIDE 136  // 272B, 16B-aligned chunks

// ---------------------------------------------------------------------------
// Tiled WMMA GEMM: C[M,N] = act(A[M,K] @ Bt[N,K]^T (+ bias[N]))
// A: bf16 row-major [M,K]; Bt: bf16 [N,K] (pre-transposed weights).
// block = 128 threads (4 waves); tile = 64(M) x 64(N); K step = 32.
// Requires K % 32 == 0, N % 64 == 0; M handled by clamping (extra rows
// computed but never stored).
// ---------------------------------------------------------------------------
template <bool RELU, bool BIAS>
__global__ __launch_bounds__(128) void gemm_wmma_kernel(
    const __bf16* __restrict__ A, const __bf16* __restrict__ Bt,
    const float* __restrict__ bias, float* __restrict__ C,
    int M, int N, int K) {
  __shared__ __attribute__((aligned(16))) __bf16 Asm[64 * GEMM_STRIDE];
  __shared__ __attribute__((aligned(16))) __bf16 Bsm[64 * GEMM_STRIDE];

  const int tid  = threadIdx.x;
  const int wave = tid >> 5;
  const int lane = tid & 31;
  const int r    = lane & 15;
  const int hi   = lane >> 4;
  const int m0   = blockIdx.y * 64;
  const int n0   = blockIdx.x * 64;

  v8f acc[4] = {};

  for (int k0 = 0; k0 < K; k0 += 32) {
#ifdef HAVE_ASYNC_LDS
    // Async stage: 64 rows x 32 bf16 = 4 x 16B chunks per row, for A and Bt.
    for (int i = tid; i < 64 * 4; i += 128) {
      const int row = i >> 2, c4 = i & 3;
      const int gm = (m0 + row < M) ? (m0 + row) : (M - 1);
      __builtin_amdgcn_global_load_async_to_lds_b128(
          ASYNC_G(A + (size_t)gm * K + k0 + c4 * 8),
          ASYNC_L(&Asm[row * GEMM_STRIDE + c4 * 8]), 0, 0);
      __builtin_amdgcn_global_load_async_to_lds_b128(
          ASYNC_G(Bt + (size_t)(n0 + row) * K + k0 + c4 * 8),
          ASYNC_L(&Bsm[row * GEMM_STRIDE + c4 * 8]), 0, 0);
    }
    wait_async_lds();
#else
    for (int i = tid; i < 64 * 32; i += 128) {
      const int row = i >> 5, col = i & 31;
      const int gm = (m0 + row < M) ? (m0 + row) : (M - 1);
      Asm[row * GEMM_STRIDE + col] = A[(size_t)gm * K + k0 + col];
      Bsm[row * GEMM_STRIDE + col] = Bt[(size_t)(n0 + row) * K + k0 + col];
    }
#endif
    __syncthreads();

    // A fragment (16x32 bf16): lanes 0-15 rows M=0..15 K{0-7,16-23},
    // lanes 16-31 same rows K{8-15,24-31}  (ISA 7.12.2).
    v16bf afrag;
    const __bf16* ar = &Asm[(wave * 16 + r) * GEMM_STRIDE + hi * 8];
#pragma unroll
    for (int p = 0; p < 8; ++p) {
      const int k = (p & 3) * 2 + ((p >> 2) ? 16 : 0);
      afrag[2 * p]     = ar[k];
      afrag[2 * p + 1] = ar[k + 1];
    }
#pragma unroll
    for (int j = 0; j < 4; ++j) {
      v16bf bfrag;
      const __bf16* br = &Bsm[(j * 16 + r) * GEMM_STRIDE + hi * 8];
#pragma unroll
      for (int p = 0; p < 8; ++p) {
        const int k = (p & 3) * 2 + ((p >> 2) ? 16 : 0);
        bfrag[2 * p]     = br[k];
        bfrag[2 * p + 1] = br[k + 1];
      }
      acc[j] = __builtin_amdgcn_wmma_f32_16x16x32_bf16(
          false, afrag, false, bfrag, (short)0, acc[j], false, false);
    }
    __syncthreads();
  }

  // D layout: lane holds col n = lane&15; VGPR v holds row v + 8*(lane>>4).
#pragma unroll
  for (int j = 0; j < 4; ++j) {
    const int gn = n0 + j * 16 + r;
#pragma unroll
    for (int v = 0; v < 8; ++v) {
      const int gm = m0 + wave * 16 + v + 8 * hi;
      if (gm < M) {
        float val = acc[j][v];
        if (BIAS) val += bias[gn];
        if (RELU) val = fmaxf(val, 0.0f);
        C[(size_t)gm * N + gn] = val;
      }
    }
  }
}

// ---------------------------------------------------------------------------
// Fused edge decoder: logits[e] = relu(concat(z[s],z[d]) @ We1 + be1) @ We2 + be2
// z: bf16 [N,64]; We1t: bf16 [64,128] (pre-transposed). Tile = 64 edges.
// Gather uses per-lane async-to-LDS copies (16B = 8 bf16 per copy).
// ---------------------------------------------------------------------------
__global__ __launch_bounds__(128) void edge_mlp_kernel(
    const __bf16* __restrict__ z, const long long* __restrict__ ei,
    const __bf16* __restrict__ We1t, const float* __restrict__ be1,
    const float* __restrict__ We2, const float* __restrict__ be2,
    float* __restrict__ out, long long E) {
  __shared__ __attribute__((aligned(16))) __bf16 Asm[64 * EMLP_STRIDE];
  __shared__ __attribute__((aligned(16))) __bf16 Bsm[64 * EMLP_STRIDE];
  __shared__ float Tsm[64 * 66];  // hidden activations [edge][hid]

  const int tid  = threadIdx.x;
  const int wave = tid >> 5;
  const int lane = tid & 31;
  const int r    = lane & 15;
  const int hi   = lane >> 4;
  const long long e0 = (long long)blockIdx.x * 64;

#ifdef HAVE_ASYNC_LDS
  // Gather A rows: 16 chunks of 8 bf16 per row; chunks 0-7 from z[src],
  // 8-15 from z[dst]. OOB edges clamp to edge 0 (rows never stored).
  for (int i = tid; i < 64 * 16; i += 128) {
    const int row = i >> 4, c8 = i & 15;
    const long long ge = (e0 + row < E) ? (e0 + row) : 0;
    const long long node = (c8 < 8) ? ei[ge] : ei[E + ge];
    __builtin_amdgcn_global_load_async_to_lds_b128(
        ASYNC_G(z + (size_t)node * 64 + (c8 & 7) * 8),
        ASYNC_L(&Asm[row * EMLP_STRIDE + c8 * 8]), 0, 0);
  }
  // Stage We1t [64][128]: 16 chunks per row.
  for (int i = tid; i < 64 * 16; i += 128) {
    const int nn = i >> 4, c8 = i & 15;
    __builtin_amdgcn_global_load_async_to_lds_b128(
        ASYNC_G(We1t + (size_t)nn * 128 + c8 * 8),
        ASYNC_L(&Bsm[nn * EMLP_STRIDE + c8 * 8]), 0, 0);
  }
  wait_async_lds();
#else
  for (int i = tid; i < 64 * 128; i += 128) {
    const int row = i >> 7, col = i & 127;
    const long long ge = (e0 + row < E) ? (e0 + row) : 0;
    const long long node = (col < 64) ? ei[ge] : ei[E + ge];
    Asm[row * EMLP_STRIDE + col] = z[(size_t)node * 64 + (col & 63)];
  }
  for (int i = tid; i < 64 * 128; i += 128) {
    const int nn = i >> 7, kk = i & 127;
    Bsm[nn * EMLP_STRIDE + kk] = We1t[(size_t)nn * 128 + kk];
  }
#endif
  __syncthreads();

  v8f acc[4] = {};
#pragma unroll
  for (int kt = 0; kt < 4; ++kt) {
    const int k0 = kt * 32;
    v16bf afrag;
    const __bf16* ar = &Asm[(wave * 16 + r) * EMLP_STRIDE + k0 + hi * 8];
#pragma unroll
    for (int p = 0; p < 8; ++p) {
      const int k = (p & 3) * 2 + ((p >> 2) ? 16 : 0);
      afrag[2 * p]     = ar[k];
      afrag[2 * p + 1] = ar[k + 1];
    }
#pragma unroll
    for (int j = 0; j < 4; ++j) {
      v16bf bfrag;
      const __bf16* br = &Bsm[(j * 16 + r) * EMLP_STRIDE + k0 + hi * 8];
#pragma unroll
      for (int p = 0; p < 8; ++p) {
        const int k = (p & 3) * 2 + ((p >> 2) ? 16 : 0);
        bfrag[2 * p]     = br[k];
        bfrag[2 * p + 1] = br[k + 1];
      }
      acc[j] = __builtin_amdgcn_wmma_f32_16x16x32_bf16(
          false, afrag, false, bfrag, (short)0, acc[j], false, false);
    }
  }

  // Hidden = relu(acc + be1) -> LDS.
#pragma unroll
  for (int j = 0; j < 4; ++j) {
    const int n = j * 16 + r;
#pragma unroll
    for (int v = 0; v < 8; ++v) {
      const int m = wave * 16 + v + 8 * hi;
      Tsm[m * 66 + n] = fmaxf(acc[j][v] + be1[n], 0.0f);
    }
  }
  __syncthreads();

  // Layer 2: 64 edges x 8 classes (K=64) via VALU.
  for (int i = tid; i < 64 * 8; i += 128) {
    const int m = i >> 3, c = i & 7;
    const long long ge = e0 + m;
    if (ge < E) {
      float s = be2[c];
#pragma unroll 8
      for (int k = 0; k < 64; ++k) s += Tsm[m * 66 + k] * We2[k * 8 + c];
      out[(size_t)ge * 8 + c] = s;
    }
  }
}

// ---------------------------------------------------------------------------
// Attention helper kernels (memory-bound scatter path, f32).
// ---------------------------------------------------------------------------
__device__ __forceinline__ void get_edge(const long long* ei, long long e,
                                         long long E, int& s, int& d) {
  if (e < E) { s = (int)ei[e]; d = (int)ei[E + e]; }
  else       { s = d = (int)(e - E); }  // appended self-loop
}

__device__ __forceinline__ float leaky02(float x) {
  return x > 0.0f ? x : 0.2f * x;
}

__device__ __forceinline__ void atomicMaxF(float* addr, float value) {
  if (value >= 0.0f) atomicMax((int*)addr, __float_as_int(value));
  else               atomicMin((unsigned int*)addr, __float_as_uint(value));
}

__global__ void node_alpha_kernel(const float* __restrict__ h,
                                  const float* __restrict__ a_s,
                                  const float* __restrict__ a_d,
                                  float* __restrict__ alpha_s,
                                  float* __restrict__ alpha_d,
                                  long long Nn, int Hh, int Cc) {
  const long long i = (long long)blockIdx.x * blockDim.x + threadIdx.x;
  if (i >= Nn * Hh) return;
  const long long n = i / Hh;
  const int hh = (int)(i % Hh);
  const float* row = h + (size_t)n * Hh * Cc + (size_t)hh * Cc;
  const float* av  = a_s + (size_t)hh * Cc;
  const float* bv  = a_d + (size_t)hh * Cc;
  float ss = 0.0f, sd = 0.0f;
  for (int c = 0; c < Cc; ++c) { const float v = row[c]; ss += v * av[c]; sd += v * bv[c]; }
  alpha_s[i] = ss;
  alpha_d[i] = sd;
}

__global__ void edge_max_kernel(const long long* __restrict__ ei,
                                const float* __restrict__ a_s,
                                const float* __restrict__ a_d,
                                float* __restrict__ emax,
                                long long E, long long Etot, int Hh) {
  const long long i = (long long)blockIdx.x * blockDim.x + threadIdx.x;
  if (i >= Etot * Hh) return;
  const long long e = i / Hh;
  const int hh = (int)(i % Hh);
  int s, d; get_edge(ei, e, E, s, d);
  const float x = leaky02(a_s[(size_t)s * Hh + hh] + a_d[(size_t)d * Hh + hh]);
  atomicMaxF(&emax[(size_t)d * Hh + hh], x);
}

__global__ void edge_den_kernel(const long long* __restrict__ ei,
                                const float* __restrict__ a_s,
                                const float* __restrict__ a_d,
                                const float* __restrict__ emax,
                                float* __restrict__ denom,
                                long long E, long long Etot, int Hh) {
  const long long i = (long long)blockIdx.x * blockDim.x + threadIdx.x;
  if (i >= Etot * Hh) return;
  const long long e = i / Hh;
  const int hh = (int)(i % Hh);
  int s, d; get_edge(ei, e, E, s, d);
  const float x = leaky02(a_s[(size_t)s * Hh + hh] + a_d[(size_t)d * Hh + hh]);
  atomicAdd(&denom[(size_t)d * Hh + hh], __expf(x - emax[(size_t)d * Hh + hh]));
}

__global__ void edge_agg_kernel(const long long* __restrict__ ei,
                                const float* __restrict__ a_s,
                                const float* __restrict__ a_d,
                                const float* __restrict__ emax,
                                const float* __restrict__ denom,
                                const float* __restrict__ feat,
                                float* __restrict__ agg,
                                long long E, long long Etot, int Hh, int Cc) {
  const long long gi = (long long)blockIdx.x * blockDim.x + threadIdx.x;
  const long long unit = gi / Cc;
  const int c = (int)(gi % Cc);
  if (unit >= Etot * Hh) return;
  const long long e = unit / Hh;
  const int hh = (int)(unit % Hh);
  int s, d; get_edge(ei, e, E, s, d);
  const float x = leaky02(a_s[(size_t)s * Hh + hh] + a_d[(size_t)d * Hh + hh]);
  const float alpha =
      __expf(x - emax[(size_t)d * Hh + hh]) / denom[(size_t)d * Hh + hh];
  atomicAdd(&agg[((size_t)d * Hh + hh) * Cc + c],
            alpha * feat[((size_t)s * Hh + hh) * Cc + c]);
}

// out_bf16 = act(in + bias): activation written directly in bf16 (feeds WMMA).
template <bool RELU>
__global__ void bias_act_bf16_kernel(const float* __restrict__ in,
                                     const float* __restrict__ bias,
                                     __bf16* __restrict__ out,
                                     long long total, int ncols) {
  const long long i = (long long)blockIdx.x * blockDim.x + threadIdx.x;
  if (i >= total) return;
  float v = in[i] + bias[i % ncols];
  if (RELU) v = fmaxf(v, 0.0f);
  out[i] = (__bf16)v;
}

__global__ void f32_to_bf16_kernel(const float* __restrict__ in,
                                   __bf16* __restrict__ out, long long n) {
  const long long i = (long long)blockIdx.x * blockDim.x + threadIdx.x;
  if (i < n) out[i] = (__bf16)in[i];
}

// Wt[n*K + k] = (bf16) W[k*N + n]
__global__ void transpose_bf16_kernel(const float* __restrict__ W,
                                      __bf16* __restrict__ Wt, int K, int N) {
  const int i = blockIdx.x * blockDim.x + threadIdx.x;
  if (i >= K * N) return;
  const int k = i / N, n = i % N;
  Wt[(size_t)n * K + k] = (__bf16)W[i];
}

__global__ void fill_kernel(unsigned int* __restrict__ p, unsigned int v,
                            long long n) {
  const long long i = (long long)blockIdx.x * blockDim.x + threadIdx.x;
  if (i < n) p[i] = v;
}

// ---------------------------------------------------------------------------
static inline long long cdiv(long long a, long long b) { return (a + b - 1) / b; }

extern "C" void kernel_launch(void* const* d_in, const int* in_sizes, int n_in,
                              void* d_out, int out_size, void* d_ws, size_t ws_size,
                              hipStream_t stream) {
  const float*     x   = (const float*)d_in[0];
  const long long* ei  = (const long long*)d_in[1];  // int64 edge_index [2,E]
  const float*     W1  = (const float*)d_in[2];
  const float*     as1 = (const float*)d_in[3];
  const float*     ad1 = (const float*)d_in[4];
  const float*     b1  = (const float*)d_in[5];
  const float*     W2  = (const float*)d_in[6];
  const float*     as2 = (const float*)d_in[7];
  const float*     ad2 = (const float*)d_in[8];
  const float*     b2  = (const float*)d_in[9];
  const float*     Wn  = (const float*)d_in[10];
  const float*     bn  = (const float*)d_in[11];
  const float*     We1 = (const float*)d_in[12];
  const float*     be1 = (const float*)d_in[13];
  const float*     We2 = (const float*)d_in[14];
  const float*     be2 = (const float*)d_in[15];

  const long long N    = in_sizes[0] / 128;  // FEAT = 128
  const long long E    = in_sizes[1] / 2;
  const long long Etot = E + N;              // with self-loops

  // Workspace layout: f32 region first (all sizes keep 16B alignment).
  float* ws    = (float*)d_ws;
  float* h1    = ws;  ws += (size_t)N * 256;  // conv1 linear (pre-aggregation)
  float* agg1  = ws;  ws += (size_t)N * 256;  // conv1 aggregation
  float* zlin  = ws;  ws += (size_t)N * 64;   // conv2 linear
  float* agg2  = ws;  ws += (size_t)N * 64;   // conv2 aggregation
  float* as1v  = ws;  ws += (size_t)N * 4;
  float* ad1v  = ws;  ws += (size_t)N * 4;
  float* emax1 = ws;  ws += (size_t)N * 4;
  float* den1  = ws;  ws += (size_t)N * 4;
  float* as2v  = ws;  ws += (size_t)N;
  float* ad2v  = ws;  ws += (size_t)N;
  float* emax2 = ws;  ws += (size_t)N;
  float* den2  = ws;  ws += (size_t)N;
  // bf16 region.
  __bf16* bws  = (__bf16*)ws;
  __bf16* xbf  = bws;  bws += (size_t)N * 128;  // x in bf16
  __bf16* hbf  = bws;  bws += (size_t)N * 256;  // relu(agg1+b1) in bf16
  __bf16* zbf  = bws;  bws += (size_t)N * 64;   // z in bf16
  __bf16* W1t  = bws;  bws += 256 * 128;        // [256][128]
  __bf16* W2t  = bws;  bws += 64 * 256;         // [64][256]
  __bf16* Wnt  = bws;  bws += 128 * 64;         // [128][64]
  __bf16* We1t = bws;  bws += 64 * 128;         // [64][128]

  float* recon = (float*)d_out;                    // [N,128]
  float* elog  = (float*)d_out + (size_t)N * 128;  // [E,8]

  const unsigned NEG_INF = 0xFF800000u;
  const int T = 256;

  // ---- init accumulators (harness does not re-poison between replays) ----
  fill_kernel<<<cdiv(N * 256, T), T, 0, stream>>>((unsigned*)agg1, 0u, N * 256);
  fill_kernel<<<cdiv(N * 4, T),   T, 0, stream>>>((unsigned*)emax1, NEG_INF, N * 4);
  fill_kernel<<<cdiv(N * 4, T),   T, 0, stream>>>((unsigned*)den1, 0u, N * 4);
  fill_kernel<<<cdiv(N * 64, T),  T, 0, stream>>>((unsigned*)agg2, 0u, N * 64);
  fill_kernel<<<cdiv(N, T),       T, 0, stream>>>((unsigned*)emax2, NEG_INF, N);
  fill_kernel<<<cdiv(N, T),       T, 0, stream>>>((unsigned*)den2, 0u, N);

  // ---- bf16 conversions (weights pre-transposed for contiguous staging) ----
  f32_to_bf16_kernel<<<cdiv(N * 128, T), T, 0, stream>>>(x, xbf, N * 128);
  transpose_bf16_kernel<<<cdiv(128 * 256, T), T, 0, stream>>>(W1, W1t, 128, 256);
  transpose_bf16_kernel<<<cdiv(256 * 64, T),  T, 0, stream>>>(W2, W2t, 256, 64);
  transpose_bf16_kernel<<<cdiv(64 * 128, T),  T, 0, stream>>>(Wn, Wnt, 64, 128);
  transpose_bf16_kernel<<<cdiv(128 * 64, T),  T, 0, stream>>>(We1, We1t, 128, 64);

  // ---- conv1: GATConv(128 -> 64, heads=4) ----
  gemm_wmma_kernel<false, false>
      <<<dim3(4, (unsigned)cdiv(N, 64)), 128, 0, stream>>>(
          xbf, W1t, nullptr, h1, (int)N, 256, 128);
  node_alpha_kernel<<<cdiv(N * 4, T), T, 0, stream>>>(h1, as1, ad1, as1v, ad1v, N, 4, 64);
  edge_max_kernel<<<cdiv(Etot * 4, T), T, 0, stream>>>(ei, as1v, ad1v, emax1, E, Etot, 4);
  edge_den_kernel<<<cdiv(Etot * 4, T), T, 0, stream>>>(ei, as1v, ad1v, emax1, den1, E, Etot, 4);
  edge_agg_kernel<<<cdiv(Etot * 4 * 64, T), T, 0, stream>>>(
      ei, as1v, ad1v, emax1, den1, h1, agg1, E, Etot, 4, 64);
  // h = relu(agg1 + b1), stored bf16 (feeds GEMM2 only)
  bias_act_bf16_kernel<true><<<cdiv(N * 256, T), T, 0, stream>>>(
      agg1, b1, hbf, N * 256, 256);

  // ---- conv2: GATConv(256 -> 64, heads=1) ----
  gemm_wmma_kernel<false, false>
      <<<dim3(1, (unsigned)cdiv(N, 64)), 128, 0, stream>>>(
          hbf, W2t, nullptr, zlin, (int)N, 64, 256);
  node_alpha_kernel<<<cdiv(N, T), T, 0, stream>>>(zlin, as2, ad2, as2v, ad2v, N, 1, 64);
  edge_max_kernel<<<cdiv(Etot, T), T, 0, stream>>>(ei, as2v, ad2v, emax2, E, Etot, 1);
  edge_den_kernel<<<cdiv(Etot, T), T, 0, stream>>>(ei, as2v, ad2v, emax2, den2, E, Etot, 1);
  edge_agg_kernel<<<cdiv(Etot * 64, T), T, 0, stream>>>(
      ei, as2v, ad2v, emax2, den2, zlin, agg2, E, Etot, 1, 64);
  // z = agg2 + b2, stored bf16 (feeds GEMM3 + edge MLP)
  bias_act_bf16_kernel<false><<<cdiv(N * 64, T), T, 0, stream>>>(
      agg2, b2, zbf, N * 64, 64);

  // ---- decode nodes: recon_x = z @ Wn + bn ----
  gemm_wmma_kernel<false, true>
      <<<dim3(2, (unsigned)cdiv(N, 64)), 128, 0, stream>>>(
          zbf, Wnt, bn, recon, (int)N, 128, 64);

  // ---- decode edges: fused gather + MLP ----
  edge_mlp_kernel<<<(unsigned)cdiv(E, 64), 128, 0, stream>>>(
      zbf, ei, We1t, be1, We2, be2, elog, E);
}
